// CAREModel_5875515261565
// MI455X (gfx1250) — matrix operand order repaired and checked by reading, not compiled, not run on hardware
//
#include <hip/hip_runtime.h>
#include <hip/hip_bf16.h>

// ---------------------------------------------------------------------------
// CDNA5 (gfx1250) CAREModel: bf16 WMMA GEMMs + double-buffered async-to-LDS.
// ---------------------------------------------------------------------------

#define N_NODES 4096
#define F_IN    300
#define F_PAD   320
#define H1      256
#define H2      128
#define C_LEN   64
#define DK      64
#define MAX_K   512
#define CAND_CAP 4096

typedef __attribute__((ext_vector_type(16))) __bf16 v16bf;
typedef __attribute__((ext_vector_type(8)))  float  v8f;

union BfFrag { v16bf v; unsigned int u[8]; };

__device__ __forceinline__ unsigned short f2bf(float f) {
    unsigned int u = __float_as_uint(f);
    u = (u + 0x7FFFu + ((u >> 16) & 1u)) >> 16;   // round-to-nearest-even
    return (unsigned short)u;
}

__device__ __forceinline__ unsigned int fsort(float f) {
    unsigned int u = __float_as_uint(f);
    return (u & 0x80000000u) ? ~u : (u | 0x80000000u);   // order-preserving
}

__device__ __forceinline__ float lrelu(float v) { return v > 0.0f ? v : 0.01f * v; }

__device__ __forceinline__ v8f wmma_bf16(const BfFrag& a, const BfFrag& b, v8f c) {
    return __builtin_amdgcn_wmma_f32_16x16x32_bf16(
        false, a.v, false, b.v, (short)0, c, false, false);
}

// Async 16-byte global -> LDS copy (ASYNCcnt-tracked), CDNA5 §08.
__device__ __forceinline__ void async_copy16(unsigned lds_addr, const void* gptr) {
    asm volatile("global_load_async_to_lds_b128 %0, %1, off"
                 :: "v"(lds_addr), "v"((unsigned long long)(size_t)gptr)
                 : "memory");
}
__device__ __forceinline__ void wait_async0() {
    asm volatile("s_wait_asynccnt 0x0" ::: "memory");
}
__device__ __forceinline__ void wait_async3() {   // allow 3 newest in flight
    asm volatile("s_wait_asynccnt 0x3" ::: "memory");
}
__device__ __forceinline__ unsigned lds_off(const void* p) {
    return (unsigned)(size_t)p;   // low 32 bits of a flat LDS address = LDS offset
}

// ---------------------------------------------------------------------------
// GEMM: C = act(A @ B),  A bf16 [M][Kpad], Bt bf16 [N][Kpad] (B transposed).
// Block tile 128x64, 8 waves (4x2), each wave 32x32 = 4 WMMA tiles per K-step.
// Double-buffered async-to-LDS pipeline (3 async ops per wave per tile).
// FLAGS: 1=lrelu, 2=store Aout bf16 [M][N], 4=store BtOut bf16 [N][btStride],
//        8=store C f32 [M][N].
// ---------------------------------------------------------------------------
#define TBM 128
#define TBN 64
#define TBK 32
#define LRS 40   // LDS row stride in halves (64B data + 16B pad)

template<int FLAGS>
__device__ __forceinline__ void store_tile(
    v8f acc, int r0, int c0, int lane, int M, int N, int btStride,
    float* __restrict__ C, unsigned short* __restrict__ Aout,
    unsigned short* __restrict__ BtOut)
{
    const int oc = c0 + (lane & 15);
    #pragma unroll
    for (int g = 0; g < 8; ++g) {
        const int orw = r0 + ((lane & 16) ? (8 + g) : g);
        if (orw < M) {
            float v = acc[g];
            if (FLAGS & 1) v = lrelu(v);
            if (FLAGS & 8) C[(size_t)orw * N + oc] = v;
            if (FLAGS & 2) Aout[(size_t)orw * N + oc] = f2bf(v);
            if (FLAGS & 4) BtOut[(size_t)oc * btStride + orw] = f2bf(v);
        }
    }
}

template<int FLAGS>
__global__ __launch_bounds__(256) void k_gemm(
    const unsigned short* __restrict__ A,   // [M][Kpad]
    const unsigned short* __restrict__ Bt,  // [N][Kpad]
    float* __restrict__ C, unsigned short* __restrict__ Aout,
    unsigned short* __restrict__ BtOut,
    int M, int N, int Kpad, int btStride)
{
    __shared__ unsigned short sA[2][TBM * LRS];
    __shared__ unsigned short sB[2][TBN * LRS];
    const int tid  = threadIdx.x;
    const int lane = tid & 31;
    const int wave = tid >> 5;
    const int wm   = wave >> 1;   // 0..3 : 32 rows each
    const int wn   = wave & 1;    // 0..1 : 32 cols each
    const int blockRow = blockIdx.x * TBM;
    const int blockCol = blockIdx.y * TBN;

    v8f acc00 = {}, acc01 = {}, acc10 = {}, acc11 = {};

    const int arow0 = tid >> 2;          // rows 0..63 (A chunk 0), +64 (chunk 1)
    const int aseg  = tid & 3;           // 16B segment within 64B row

    auto issue = [&](int k0, int buf) {
        #pragma unroll
        for (int r = 0; r < 2; ++r) {
            const int row = arow0 + r * 64;
            int gr = blockRow + row; if (gr >= M) gr = M - 1;
            async_copy16(lds_off(&sA[buf][row * LRS + aseg * 8]),
                         A + (size_t)gr * Kpad + k0 + aseg * 8);
        }
        int gc = blockCol + arow0; if (gc >= N) gc = N - 1;
        async_copy16(lds_off(&sB[buf][arow0 * LRS + aseg * 8]),
                     Bt + (size_t)gc * Kpad + k0 + aseg * 8);
    };

    const int kTiles = Kpad / TBK;
    issue(0, 0);
    for (int kt = 0; kt < kTiles; ++kt) {
        const int buf = kt & 1;
        if (kt + 1 < kTiles) {                 // prefetch next tile, then wait
            issue((kt + 1) * TBK, buf ^ 1);    // for this tile's 3 oldest ops
            wait_async3();
        } else {
            wait_async0();
        }
        __syncthreads();

        const int mrow = lane & 15;
        const int kb   = (lane >> 4) * 8;    // halves (unused)
        const int kb2  = (lane >> 4) * 16;   // halves (B frag)
        BfFrag a0, a1, b0, b1;
        const unsigned int* ap0 = (const unsigned int*)&sA[buf][(wm * 32 + mrow) * LRS];
        const unsigned int* ap1 = (const unsigned int*)&sA[buf][(wm * 32 + 16 + mrow) * LRS];
        #pragma unroll
        for (int i = 0; i < 4; ++i) {
            a0.u[i] = ap0[(kb2 >> 1) + i];  a0.u[4 + i] = ap0[4 + (kb2 >> 1) + i];
            a1.u[i] = ap1[(kb2 >> 1) + i];  a1.u[4 + i] = ap1[4 + (kb2 >> 1) + i];
        }
        const unsigned int* bp0 = (const unsigned int*)&sB[buf][(wn * 32 + mrow) * LRS + kb2];
        const unsigned int* bp1 = (const unsigned int*)&sB[buf][(wn * 32 + 16 + mrow) * LRS + kb2];
        #pragma unroll
        for (int i = 0; i < 8; ++i) { b0.u[i] = bp0[i]; b1.u[i] = bp1[i]; }

        acc00 = wmma_bf16(a0, b0, acc00);
        acc01 = wmma_bf16(a0, b1, acc01);
        acc10 = wmma_bf16(a1, b0, acc10);
        acc11 = wmma_bf16(a1, b1, acc11);
        __syncthreads();
    }

    const int r0 = blockRow + wm * 32;
    const int c0 = blockCol + wn * 32;
    store_tile<FLAGS>(acc00, r0,      c0,      lane, M, N, btStride, C, Aout, BtOut);
    store_tile<FLAGS>(acc01, r0,      c0 + 16, lane, M, N, btStride, C, Aout, BtOut);
    store_tile<FLAGS>(acc10, r0 + 16, c0,      lane, M, N, btStride, C, Aout, BtOut);
    store_tile<FLAGS>(acc11, r0 + 16, c0 + 16, lane, M, N, btStride, C, Aout, BtOut);
}

// ---------------------------------------------------------------------------
// z @ z^T tile (16x16 logits per wave) via WMMA, zb = bf16 rows of 128.
// ---------------------------------------------------------------------------
__device__ __forceinline__ v8f zz_tile(const unsigned short* __restrict__ zb,
                                       int rowBase, int colBase, int lane)
{
    v8f acc = {};
    const int m   = rowBase + (lane & 15);
    const int nc  = colBase + (lane & 15);
    const int kb  = (lane >> 4) * 8;
    const int kb2 = (lane >> 4) * 16;
    const unsigned int* za  = (const unsigned int*)(zb + (size_t)m  * H2);
    const unsigned int* zbp = (const unsigned int*)(zb + (size_t)nc * H2);
    #pragma unroll
    for (int ks = 0; ks < 4; ++ks) {                  // K = 128 = 4 * 32
        BfFrag a, b;
        const int base = ks * 16;                     // uints (32 halves)
        #pragma unroll
        for (int i = 0; i < 4; ++i) {
            a.u[i]     = za[base + (kb2 >> 1) + i];
            a.u[4 + i] = za[base + 4 + (kb2 >> 1) + i];
        }
        #pragma unroll
        for (int i = 0; i < 8; ++i) b.u[i] = zbp[base + (kb2 >> 1) + i];
        acc = wmma_bf16(a, b, acc);
    }
    return acc;
}

// Fused loss pass: BCE over sigmoid(z z^T) + histogram of triu logits.
__global__ __launch_bounds__(256) void k_loss(
    const unsigned short* __restrict__ zb, const float* __restrict__ labels,
    float* __restrict__ sc, unsigned int* __restrict__ hist)
{
    __shared__ unsigned int lh[1024];
    __shared__ float red[256];
    const int tid  = threadIdx.x;
    const int lane = tid & 31;
    const int wave = tid >> 5;
    for (int i = tid; i < 1024; i += 256) lh[i] = 0u;
    __syncthreads();

    const int tr = blockIdx.x * 64 + (wave >> 1) * 16;
    const int tc = blockIdx.y * 32 + (wave & 1) * 16;
    v8f c = zz_tile(zb, tr, tc, lane);

    const float nf = (float)N_NODES;
    const float nn = nf * nf;
    const float ls = sc[0];
    const float pw = (nn - ls + nf) / (ls - nf + 0.01f);

    float bsum = 0.0f;
    const int j = tc + (lane & 15);
    #pragma unroll
    for (int g = 0; g < 8; ++g) {
        const int i = tr + ((lane & 16) ? (8 + g) : g);
        const float logit = c[g];
        const float x   = 1.0f / (1.0f + __expf(-logit));  // recover_adj
        const float lab = labels[(size_t)i * N_NODES + j];
        const float lsx  = -log1pf(__expf(-x));            // log_sigmoid(x)
        const float lsnx = -log1pf(__expf(x));             // log_sigmoid(-x)
        bsum += -(pw * lab * lsx + (1.0f - lab) * lsnx);
        if (j > i) atomicAdd(&lh[fsort(logit) >> 22], 1u);
    }
    red[tid] = bsum;
    __syncthreads();
    for (int s = 128; s > 0; s >>= 1) {
        if (tid < s) red[tid] += red[tid + s];
        __syncthreads();
    }
    if (tid == 0) atomicAdd(&sc[1], red[0]);
    __syncthreads();
    for (int i = tid; i < 1024; i += 256)
        if (lh[i]) atomicAdd(&hist[i], lh[i]);
}

__global__ void k_threshold(const unsigned int* __restrict__ hist,
                            unsigned int* __restrict__ thr)
{
    if (threadIdx.x == 0 && blockIdx.x == 0) {
        unsigned int cum = 0; int T = 0;
        for (int b = 1023; b >= 0; --b) {
            cum += hist[b];
            if (cum >= (unsigned int)MAX_K) { T = b; break; }
        }
        *thr = ((unsigned int)T) << 22;
    }
}

__global__ __launch_bounds__(256) void k_collect(
    const unsigned short* __restrict__ zb, const unsigned int* __restrict__ thr_p,
    unsigned int* __restrict__ counter, unsigned int* __restrict__ ckey,
    unsigned int* __restrict__ cidx)
{
    const int tid  = threadIdx.x;
    const int lane = tid & 31;
    const int wave = tid >> 5;
    const int tr = blockIdx.x * 64 + (wave >> 1) * 16;
    const int tc = blockIdx.y * 32 + (wave & 1) * 16;
    v8f c = zz_tile(zb, tr, tc, lane);
    const unsigned int thr = *thr_p;
    const int j = tc + (lane & 15);
    #pragma unroll
    for (int g = 0; g < 8; ++g) {
        const int i = tr + ((lane & 16) ? (8 + g) : g);
        if (j > i) {
            const unsigned int u = fsort(c[g]);
            if (u >= thr) {
                unsigned int pos = atomicAdd(counter, 1u);
                if (pos < (unsigned int)CAND_CAP) {
                    ckey[pos] = u;
                    cidx[pos] = (unsigned int)(i * N_NODES + j);
                }
            }
        }
    }
}

__global__ __launch_bounds__(256) void k_select(
    const unsigned int* __restrict__ ckey, const unsigned int* __restrict__ cidx,
    const unsigned int* __restrict__ cnt_p, const float* __restrict__ m,
    float* __restrict__ out_rel)
{
    __shared__ unsigned int skey[CAND_CAP];
    __shared__ unsigned int rkey[256];
    __shared__ unsigned int rpos[256];
    __shared__ unsigned int sel[MAX_K];
    const int tid = threadIdx.x;
    unsigned int cnt = *cnt_p;
    if (cnt > (unsigned int)CAND_CAP) cnt = CAND_CAP;
    for (int i = tid; i < CAND_CAP; i += 256)
        skey[i] = (i < (int)cnt) ? ckey[i] : 0u;
    __syncthreads();

    const int nsel = ((int)cnt < MAX_K) ? (int)cnt : MAX_K;
    for (int r = 0; r < MAX_K; ++r) {
        unsigned int bk = 0u, bp = 0u;
        for (int i = tid; i < CAND_CAP; i += 256)
            if (skey[i] > bk) { bk = skey[i]; bp = (unsigned int)i; }
        rkey[tid] = bk; rpos[tid] = bp;
        __syncthreads();
        for (int s = 128; s > 0; s >>= 1) {
            if (tid < s && rkey[tid + s] > rkey[tid]) {
                rkey[tid] = rkey[tid + s]; rpos[tid] = rpos[tid + s];
            }
            __syncthreads();
        }
        if (tid == 0) { sel[r] = rpos[0]; skey[rpos[0]] = 0u; }
        __syncthreads();
    }
    for (int t = tid; t < MAX_K * H2; t += 256) {
        const int r = t >> 7, c = t & 127;
        float v = 0.0f;
        if (r < nsel) {
            const unsigned int idx = cidx[sel[r]];
            const int i = (int)(idx >> 12);       // N = 4096
            const int j = (int)(idx & 4095u);
            v = m[(size_t)i * H2 + c] + m[(size_t)j * H2 + c];
        }
        out_rel[t] = v;
    }
}

// ---------------------------------------------------------------------------
// Attention: one thread per (row, head); writes bf16 O for the Wo GEMM.
// ---------------------------------------------------------------------------
__global__ __launch_bounds__(256) void k_attn(
    const float* __restrict__ Q, const float* __restrict__ Kc,
    const float* __restrict__ Vc, unsigned short* __restrict__ O)
{
    const int t = blockIdx.x * 256 + threadIdx.x;
    if (t >= N_NODES * 2) return;
    const int row = t >> 1;
    const int head = t & 1;
    const float* q = Q + (size_t)row * DK + head * 32;
    float e[C_LEN];
    float mx = -1e30f;
    for (int c = 0; c < C_LEN; ++c) {
        const float* kr = Kc + (size_t)c * DK + head * 32;
        float s = 0.0f;
        #pragma unroll
        for (int d = 0; d < 32; ++d) s += q[d] * kr[d];
        s *= 0.1767766953f;              // 1/sqrt(32)
        e[c] = s;
        mx = fmaxf(mx, s);
    }
    float den = 0.0f;
    for (int c = 0; c < C_LEN; ++c) { e[c] = __expf(e[c] - mx); den += e[c]; }
    const float inv = 1.0f / den;
    for (int d = 0; d < 32; ++d) {
        float a = 0.0f;
        for (int c = 0; c < C_LEN; ++c) a += e[c] * Vc[(size_t)c * DK + head * 32 + d];
        O[(size_t)row * DK + head * 32 + d] = f2bf(a * inv);
    }
}

// ---------------------------------------------------------------------------
// Conversions and elementwise helpers.
// ---------------------------------------------------------------------------
__global__ __launch_bounds__(256) void k_f2bf(const float* __restrict__ src,
                                              unsigned short* __restrict__ dst,
                                              int R, int C, int Cpad)
{
    for (size_t t = (size_t)blockIdx.x * 256 + threadIdx.x; t < (size_t)R * Cpad;
         t += (size_t)gridDim.x * 256) {
        const int r = (int)(t / Cpad), c = (int)(t % Cpad);
        dst[t] = (c < C) ? f2bf(src[(size_t)r * C + c]) : (unsigned short)0;
    }
}

// dst[n][Kpad] = src[k][n]  (weights -> transposed bf16)
__global__ __launch_bounds__(256) void k_f2bf_T(const float* __restrict__ src,
                                                unsigned short* __restrict__ dst,
                                                int K, int N, int Kpad)
{
    for (size_t t = (size_t)blockIdx.x * 256 + threadIdx.x; t < (size_t)N * Kpad;
         t += (size_t)gridDim.x * 256) {
        const int n = (int)(t / Kpad), k = (int)(t % Kpad);
        dst[t] = (k < K) ? f2bf(src[(size_t)k * N + n]) : (unsigned short)0;
    }
}

__global__ __launch_bounds__(256) void k_sum(const float* __restrict__ x, int n,
                                             float* __restrict__ dst)
{
    __shared__ float red[256];
    float acc = 0.0f;
    for (size_t t = (size_t)blockIdx.x * 256 + threadIdx.x; t < (size_t)n;
         t += (size_t)gridDim.x * 256) acc += x[t];
    red[threadIdx.x] = acc;
    __syncthreads();
    for (int s = 128; s > 0; s >>= 1) {
        if ((int)threadIdx.x < s) red[threadIdx.x] += red[threadIdx.x + s];
        __syncthreads();
    }
    if (threadIdx.x == 0) atomicAdd(dst, red[0]);
}

__global__ __launch_bounds__(256) void k_z(const float* __restrict__ muv,
                                           const float* __restrict__ eps,
                                           unsigned short* __restrict__ zb)
{
    const int t = blockIdx.x * 256 + threadIdx.x;
    if (t >= N_NODES * H2) return;
    const int r = t >> 7, c = t & 127;
    const float mu = muv[(size_t)r * 256 + c];
    const float lv = muv[(size_t)r * 256 + 128 + c];
    zb[t] = f2bf(eps[t] * __expf(0.5f * lv) + mu);
}

__global__ __launch_bounds__(256) void k_kld(const float* __restrict__ post,
                                             const float* __restrict__ prior,
                                             float* __restrict__ sc)
{
    __shared__ float red[256];
    float acc = 0.0f;
    for (int t = blockIdx.x * 256 + threadIdx.x; t < N_NODES * H2;
         t += gridDim.x * 256) {
        const int r = t >> 7, c = t & 127;
        const float mu  = post[(size_t)r * 256 + c];
        const float lv  = post[(size_t)r * 256 + 128 + c];
        const float mup = prior[(size_t)r * 256 + c];
        const float lvp = prior[(size_t)r * 256 + 128 + c];
        const float d = mup - mu;
        acc += d * d * __expf(-lvp) + __expf(lv - lvp) - 1.0f - (lv - lvp);
    }
    red[threadIdx.x] = acc;
    __syncthreads();
    for (int s = 128; s > 0; s >>= 1) {
        if ((int)threadIdx.x < s) red[threadIdx.x] += red[threadIdx.x + s];
        __syncthreads();
    }
    if (threadIdx.x == 0) atomicAdd(&sc[2], red[0]);
}

__global__ void k_init(float* __restrict__ sc, unsigned int* __restrict__ hist,
                       unsigned int* __restrict__ counter,
                       unsigned int* __restrict__ thr)
{
    const int t = blockIdx.x * 256 + threadIdx.x;
    if (t < 16)   sc[t] = 0.0f;
    if (t < 1024) hist[t] = 0u;
    if (t == 0) { *counter = 0u; *thr = 0u; }
}

__global__ void k_final(const float* __restrict__ sc, float* __restrict__ out)
{
    const int tid = threadIdx.x;
    if (tid < MAX_K) out[MAX_K * H2 + tid] = 0.0f;   // rel_mask: all False
    if (tid == 0) {
        const float nf = (float)N_NODES;
        const float nn = nf * nf;
        const float ls = sc[0];
        const float norm = nn / (nn - ls + nf);
        out[MAX_K * H2 + MAX_K]     = norm * sc[1] / nn;     // recons_loss
        out[MAX_K * H2 + MAX_K + 1] = 0.5f * sc[2] / nn;     // kld
    }
}

// ---------------------------------------------------------------------------
// Host orchestration.
// ---------------------------------------------------------------------------
static void launch_gemm(const unsigned short* A, const unsigned short* Bt,
                        float* C, unsigned short* Aout, unsigned short* BtOut,
                        int M, int N, int Kpad, int btStride, int flags,
                        hipStream_t stream)
{
    dim3 grid((M + TBM - 1) / TBM, N / TBN);
    dim3 blk(256);
    switch (flags) {
    case 2: k_gemm<2><<<grid, blk, 0, stream>>>(A, Bt, C, Aout, BtOut, M, N, Kpad, btStride); break;
    case 3: k_gemm<3><<<grid, blk, 0, stream>>>(A, Bt, C, Aout, BtOut, M, N, Kpad, btStride); break;
    case 5: k_gemm<5><<<grid, blk, 0, stream>>>(A, Bt, C, Aout, BtOut, M, N, Kpad, btStride); break;
    case 8: k_gemm<8><<<grid, blk, 0, stream>>>(A, Bt, C, Aout, BtOut, M, N, Kpad, btStride); break;
    default: k_gemm<9><<<grid, blk, 0, stream>>>(A, Bt, C, Aout, BtOut, M, N, Kpad, btStride); break;
    }
}

extern "C" void kernel_launch(void* const* d_in, const int* in_sizes, int n_in,
                              void* d_out, int out_size, void* d_ws, size_t ws_size,
                              hipStream_t stream)
{
    (void)in_sizes; (void)n_in; (void)out_size; (void)ws_size;
    const float* ns_emb = (const float*)d_in[0];
    const float* adj    = (const float*)d_in[1];
    const float* adj_pr = (const float*)d_in[2];
    const float* cond   = (const float*)d_in[3];
    const float* labels = (const float*)d_in[4];
    const float* eps    = (const float*)d_in[5];
    const float* W[15];
    for (int i = 0; i < 15; ++i) W[i] = (const float*)d_in[6 + i];
    float* out = (float*)d_out;

    // ---- workspace carve-up ----
    float* w = (float*)d_ws;
    float*        sc      = w;
    unsigned int* counter = (unsigned int*)(w + 16);
    unsigned int* thr     = (unsigned int*)(w + 17);
    unsigned int* hist    = (unsigned int*)(w + 64);
    unsigned int* ckey    = (unsigned int*)(w + 2048);
    unsigned int* cidx    = (unsigned int*)(w + 2048 + CAND_CAP);

    unsigned short* ub = (unsigned short*)(w + 16384);
    size_t uo = 0;
    auto ualloc = [&](size_t n) { unsigned short* p = ub + uo; uo += n; return p; };
    unsigned short* adj_b  = ualloc((size_t)N_NODES * N_NODES);
    unsigned short* adjp_b = ualloc((size_t)N_NODES * N_NODES);
    unsigned short* x_b    = ualloc((size_t)N_NODES * F_PAD);
    unsigned short* cond_b = ualloc((size_t)C_LEN * H1);
    unsigned short* wT[15];
    const int wK[15]  = {F_IN, H1, H1, H1, DK, H1, H1,  F_IN, H1, H1, H1, DK, H1, H1,  F_IN};
    const int wKp[15] = {F_PAD, H1, H1, H1, DK, H1, H1, F_PAD, H1, H1, H1, DK, H1, H1, F_PAD};
    const int wN[15]  = {H1, DK, DK, DK, H1, H2, H2,    H1, DK, DK, DK, H1, H2, H2,   H2};
    for (int i = 0; i < 15; ++i) wT[i] = ualloc((size_t)wN[i] * wKp[i]);
    unsigned short* s_bt  = ualloc((size_t)H1 * N_NODES);   // [256][4096]
    unsigned short* t_bt  = ualloc((size_t)H1 * N_NODES);   // [256][4096]
    unsigned short* h_ab  = ualloc((size_t)N_NODES * H1);
    unsigned short* h2_ab = ualloc((size_t)N_NODES * H1);
    unsigned short* o_ab  = ualloc((size_t)N_NODES * DK);
    unsigned short* zb    = ualloc((size_t)N_NODES * H2);
    if (uo & 1) ++uo;                                       // 4B-align the floats
    float* fb = (float*)(ub + uo);
    size_t fo = 0;
    auto falloc = [&](size_t n) { float* p = fb + fo; fo += n; return p; };
    float* q_buf    = falloc((size_t)N_NODES * DK);
    float* k_buf    = falloc((size_t)C_LEN * DK);
    float* v_buf    = falloc((size_t)C_LEN * DK);
    float* muv_post = falloc((size_t)N_NODES * H1);
    float* muv_pri  = falloc((size_t)N_NODES * H1);
    float* m_buf    = falloc((size_t)N_NODES * H2);

    auto gemm = [&](const unsigned short* A, const unsigned short* Bt,
                    float* C, unsigned short* Aout, unsigned short* BtOut,
                    int M, int N, int Kpad, int btStride, int flags) {
        launch_gemm(A, Bt, C, Aout, BtOut, M, N, Kpad, btStride, flags, stream);
    };
    auto cvt = [&](const float* s, unsigned short* d, int R, int C, int Cp) {
        size_t n = (size_t)R * Cp;
        int g = (int)((n + 255) / 256); if (g > 4096) g = 4096;
        k_f2bf<<<dim3(g), dim3(256), 0, stream>>>(s, d, R, C, Cp);
    };

    k_init<<<dim3(4), dim3(256), 0, stream>>>(sc, hist, counter, thr);
    k_sum<<<dim3(1024), dim3(256), 0, stream>>>(labels, N_NODES * N_NODES, &sc[0]);

    // ---- one-time bf16 conversions ----
    cvt(ns_emb, x_b, N_NODES, F_IN, F_PAD);
    cvt(adj,    adj_b,  N_NODES, N_NODES, N_NODES);
    cvt(adj_pr, adjp_b, N_NODES, N_NODES, N_NODES);
    cvt(cond,   cond_b, C_LEN, H1, H1);
    for (int i = 0; i < 15; ++i) {
        size_t n = (size_t)wN[i] * wKp[i];
        int g = (int)((n + 255) / 256); if (g > 1024) g = 1024;
        k_f2bf_T<<<dim3(g), dim3(256), 0, stream>>>(W[i], wT[i], wK[i], wN[i], wKp[i]);
    }

    // ---- encoders (posterior enc=0, prior enc=1) ----
    for (int enc = 0; enc < 2; ++enc) {
        const unsigned short* Ab = enc ? adjp_b : adj_b;
        unsigned short* const* Wt = enc ? &wT[7] : &wT[0];
        float* muv = enc ? muv_pri : muv_post;
        // gcn1: h = lrelu(adj @ lrelu(x @ Wh))
        gemm(x_b, Wt[0], nullptr, nullptr, s_bt, N_NODES, H1, F_PAD, N_NODES, 1 | 4);
        gemm(Ab, s_bt, nullptr, h_ab, nullptr, N_NODES, H1, N_NODES, 0, 1 | 2);
        // mha
        gemm(h_ab,   Wt[1], q_buf, nullptr, nullptr, N_NODES, DK, H1, 0, 8);
        gemm(cond_b, Wt[2], k_buf, nullptr, nullptr, C_LEN,   DK, H1, 0, 8);
        gemm(cond_b, Wt[3], v_buf, nullptr, nullptr, C_LEN,   DK, H1, 0, 8);
        k_attn<<<dim3((N_NODES * 2 + 255) / 256), dim3(256), 0, stream>>>(
            q_buf, k_buf, v_buf, o_ab);
        gemm(o_ab, Wt[4], nullptr, h2_ab, nullptr, N_NODES, H1, DK, 0, 2);
        // mu / logvar inner GEMMs -> transposed-concat [mu|logvar] as Bt
        gemm(h2_ab, Wt[5], nullptr, nullptr, t_bt,                        N_NODES, H2, H1, N_NODES, 1 | 4);
        gemm(h2_ab, Wt[6], nullptr, nullptr, t_bt + (size_t)H2 * N_NODES, N_NODES, H2, H1, N_NODES, 1 | 4);
        gemm(Ab, t_bt, muv, nullptr, nullptr, N_NODES, 2 * H2, N_NODES, 0, 1 | 8);
    }

    // ---- node mapping, kld, z ----
    gemm(x_b, wT[14], m_buf, nullptr, nullptr, N_NODES, H2, F_PAD, 0, 1 | 8);
    k_kld<<<dim3(512), dim3(256), 0, stream>>>(muv_post, muv_pri, sc);
    k_z<<<dim3((N_NODES * H2 + 255) / 256), dim3(256), 0, stream>>>(muv_post, eps, zb);

    // ---- fused z z^T: BCE + histogram, threshold, collect, select ----
    dim3 zgrid(N_NODES / 64, N_NODES / 32);
    k_loss<<<zgrid, dim3(256), 0, stream>>>(zb, labels, sc, hist);
    k_threshold<<<dim3(1), dim3(64), 0, stream>>>(hist, thr);
    k_collect<<<zgrid, dim3(256), 0, stream>>>(zb, thr, counter, ckey, cidx);
    k_select<<<dim3(1), dim3(256), 0, stream>>>(ckey, cidx, counter, m_buf, out);

    k_final<<<dim3(1), dim3(512), 0, stream>>>(sc, out);
}
